// AttenBlock_87170656240331
// MI455X (gfx1250) — compile-verified, matrix operand order checked by reading, not run on hardware
//
#include <hip/hip_runtime.h>
#include <hip/hip_bf16.h>
#include <math.h>

typedef __attribute__((ext_vector_type(16))) _Float16 v16h;
typedef __attribute__((ext_vector_type(8)))  float    v8f;

#define DEVFN static __device__ __forceinline__

DEVFN float silu_f(float x)     { return x / (1.f + __expf(-x)); }
DEVFN float gelu_f(float x)     { return 0.5f * x * (1.f + erff(x * 0.70710678118654752f)); }
DEVFN float softplus_f(float x) { return (x > 20.f) ? x : log1pf(__expf(x)); }

// ---------------------------------------------------------------------------
// WMMA GEMM:  Y[M,N] = X[M,K] * W[N,K]^T (+ bias) (+= Y if accumulate)
//
// Block = 128 threads = 4 waves; each wave owns one 16-row M tile, the block
// shares one 32-wide N tile pair. Per 32-deep k-step the block stages the
// 32x32 f32 B chunk into LDS:
//   - K%8==0 (64/128/256): two global_load_async_to_lds_b128 per thread
//     (ASYNCcnt path, VGPR-free), s_wait_asynccnt 0 before the barrier.
//   - otherwise: float2 loads + LDS stores (K=170), masked scalars at edges.
// Waves then read 16B-aligned float4 runs from LDS, convert to f16, and issue
// two v_wmma_f32_16x16x32_f16.
// f16 A layout: lane owns row (lane&15); lanes<16 hold K runs [0..7],[16..23],
// lanes>=16 hold [8..15],[24..31] -> two contiguous 8-float spans (b64 loads).
// ---------------------------------------------------------------------------
DEVFN v16h load16_fast(const float* __restrict__ row, int k0, int koff) {
  const float2* p0 = (const float2*)(row + k0 + koff);
  const float2* p1 = (const float2*)(row + k0 + koff + 16);
  v16h r;
#pragma unroll
  for (int q = 0; q < 4; ++q) {
    float2 t = p0[q];
    r[q * 2]     = (_Float16)t.x;
    r[q * 2 + 1] = (_Float16)t.y;
  }
#pragma unroll
  for (int q = 0; q < 4; ++q) {
    float2 t = p1[q];
    r[8 + q * 2]     = (_Float16)t.x;
    r[8 + q * 2 + 1] = (_Float16)t.y;
  }
  return r;
}

DEVFN v16h load16_masked(const float* __restrict__ row, int k0, int koff, int K) {
  v16h r;
#pragma unroll
  for (int e = 0; e < 16; ++e) {
    const int kl = ((e < 8) ? e : (e + 8)) + koff;
    const int kk = k0 + kl;
    r[e] = (_Float16)((kk < K) ? row[kk] : 0.f);
  }
  return r;
}

DEVFN void pack8(v16h& d, int base, float4 lo, float4 hi) {
  d[base + 0] = (_Float16)lo.x; d[base + 1] = (_Float16)lo.y;
  d[base + 2] = (_Float16)lo.z; d[base + 3] = (_Float16)lo.w;
  d[base + 4] = (_Float16)hi.x; d[base + 5] = (_Float16)hi.y;
  d[base + 6] = (_Float16)hi.z; d[base + 7] = (_Float16)hi.w;
}

__global__ void gemm16_wmma(const float* __restrict__ X, const float* __restrict__ W,
                            const float* __restrict__ bias, float* __restrict__ Y,
                            int M, int N, int K, int accumulate) {
  __shared__ float Bsf[32 * 32];         // [n 0..31][k 0..31] f32, 4 KB
                                         // (only LDS object -> LDS offset 0)
  const int tileN0 = blockIdx.x * 2;     // two 16-wide N tiles per block
  const int tid  = threadIdx.x;          // 0..127
  const int wid  = tid >> 5;
  const int lane = tid & 31;
  const int tileM = blockIdx.y * 4 + wid;
  const int mh   = lane & 15;
  const int koff = (lane < 16) ? 0 : 8;
  const int mrow = tileM * 16 + mh;      // M is a multiple of 64 here
  const float* xrow = X + (size_t)mrow * K;

  // B loader coords: thread stages 8 k-values of one n row
  const int ln = tid >> 2;               // 0..31 : n within tile pair
  const int lk = (tid & 3) * 8;          // 0,8,16,24
  const int gn = tileN0 * 16 + ln;
  const bool nok = (gn < N);
  const float* wrow = W + (size_t)(nok ? gn : 0) * K;
  const unsigned ldsOff = (unsigned)((ln * 32 + lk) * 4);   // byte offset in LDS
  const bool asyncOK = ((K & 7) == 0);   // 16B-aligned rows -> b128 async legal

  v8f acc0 = {}, acc1 = {};
  for (int k0 = 0; k0 < K; k0 += 32) {
    __syncthreads();
    if (nok && (k0 + 32 <= K) && asyncOK) {
      // async global -> LDS copy, 32B per thread (offset applies to both sides)
      const unsigned long long ga = (unsigned long long)(const void*)(wrow + k0 + lk);
      asm volatile("global_load_async_to_lds_b128 %0, %1, off"
                   :: "v"(ldsOff), "v"(ga) : "memory");
      asm volatile("global_load_async_to_lds_b128 %0, %1, off offset:16"
                   :: "v"(ldsOff), "v"(ga) : "memory");
      asm volatile("s_wait_asynccnt 0" ::: "memory");
    } else if (nok && (k0 + 32 <= K)) {  // K even -> 8B-aligned float2 loads
      const float2* p = (const float2*)(wrow + k0 + lk);
#pragma unroll
      for (int q = 0; q < 4; ++q) {
        float2 t = p[q];
        Bsf[ln * 32 + lk + q * 2]     = t.x;
        Bsf[ln * 32 + lk + q * 2 + 1] = t.y;
      }
    } else {
#pragma unroll
      for (int q = 0; q < 8; ++q) {
        const int kk = k0 + lk + q;
        Bsf[ln * 32 + lk + q] = (nok && kk < K) ? wrow[kk] : 0.f;
      }
    }
    __syncthreads();

    v16h a;
    if (k0 + 32 <= K) {
      __builtin_prefetch(xrow + k0 + 32, 0, 3);
      a = load16_fast(xrow, k0, koff);
    } else {
      a = load16_masked(xrow, k0, koff, K);
    }

    // B from LDS: float4 runs at 16B-aligned offsets, convert to f16
    const float4* bp = (const float4*)Bsf;
    const int r0 = (lane & 15) * 8 + (koff >> 2);   // (n*32+koff)/4
    const int r1 = r0 + 16 * 8;                     // n + 16
    v16h b0, b1;
    pack8(b0, 0, bp[r0],     bp[r0 + 1]);
    pack8(b0, 8, bp[r0 + 4], bp[r0 + 5]);
    pack8(b1, 0, bp[r1],     bp[r1 + 1]);
    pack8(b1, 8, bp[r1 + 4], bp[r1 + 5]);
    acc0 = __builtin_amdgcn_wmma_f32_16x16x32_f16(false, a, false, b0, (short)0, acc0, false, false);
    acc1 = __builtin_amdgcn_wmma_f32_16x16x32_f16(false, a, false, b1, (short)0, acc1, false, false);
  }

  // C/D layout: vgpr r -> M = r + (lane<16 ? 0 : 8), N = lane&15
  const int mbase = tileM * 16 + ((lane < 16) ? 0 : 8);
  const int n0 = tileN0 * 16 + mh;
  const int n1 = n0 + 16;
  if (n0 < N) {
    const float bv = bias ? bias[n0] : 0.f;
#pragma unroll
    for (int r = 0; r < 8; ++r) {
      const size_t off = (size_t)(mbase + r) * N + n0;
      float v = acc0[r] + bv;
      if (accumulate) v += Y[off];
      Y[off] = v;
    }
  }
  if (n1 < N) {
    const float bv = bias ? bias[n1] : 0.f;
#pragma unroll
    for (int r = 0; r < 8; ++r) {
      const size_t off = (size_t)(mbase + r) * N + n1;
      float v = acc1[r] + bv;
      if (accumulate) v += Y[off];
      Y[off] = v;
    }
  }
}

// ---------------------------------------------------------------------------
// LayerNorm over last dim, one wave per row, eps = 1e-5 (population variance)
// ---------------------------------------------------------------------------
__global__ void ln_kernel(const float* __restrict__ X, const float* __restrict__ g,
                          const float* __restrict__ b, float* __restrict__ Y, int C) {
  const size_t row = blockIdx.x;
  const int lane = threadIdx.x;
  const float* xr = X + row * C;
  float s = 0.f, s2 = 0.f;
  for (int c = lane; c < C; c += 32) { float v = xr[c]; s += v; s2 += v * v; }
#pragma unroll
  for (int m = 16; m; m >>= 1) { s += __shfl_xor(s, m, 32); s2 += __shfl_xor(s2, m, 32); }
  const float mu  = s / C;
  const float var = s2 / C - mu * mu;
  const float inv = rsqrtf(var + 1e-5f);
  float* yr = Y + row * C;
  for (int c = lane; c < C; c += 32) yr[c] = (xr[c] - mu) * inv * g[c] + b[c];
}

// ---------------------------------------------------------------------------
// Depthwise 3x3, SAME, channel-last [B,H,W,C]; weight [C,1,3,3]
// ---------------------------------------------------------------------------
__global__ void dwconv3_kernel(const float* __restrict__ X, const float* __restrict__ Wt,
                               const float* __restrict__ bias, float* __restrict__ Y,
                               int Bn, int H, int Wd, int C) {
  const size_t idx = (size_t)blockIdx.x * blockDim.x + threadIdx.x;
  const size_t total = (size_t)Bn * H * Wd * C;
  if (idx >= total) return;
  const int c = (int)(idx % C);
  size_t r = idx / C;
  const int j = (int)(r % Wd); r /= Wd;
  const int i = (int)(r % H);
  const int bb = (int)(r / H);
  float acc = bias ? bias[c] : 0.f;
#pragma unroll
  for (int di = -1; di <= 1; ++di)
#pragma unroll
    for (int dj = -1; dj <= 1; ++dj) {
      const int ii = i + di, jj = j + dj;
      if (ii < 0 || ii >= H || jj < 0 || jj >= Wd) continue;
      acc += X[(((size_t)bb * H + ii) * Wd + jj) * C + c] * Wt[c * 9 + (di + 1) * 3 + (dj + 1)];
    }
  Y[idx] = acc;
}

// ---------------------------------------------------------------------------
// Elementwise helpers
// ---------------------------------------------------------------------------
__global__ void copy_kernel(const float* __restrict__ S, float* __restrict__ D, size_t n) {
  size_t i = (size_t)blockIdx.x * blockDim.x + threadIdx.x;
  if (i < n) D[i] = S[i];
}
__global__ void silu_inplace_kernel(float* __restrict__ P, size_t n) {
  size_t i = (size_t)blockIdx.x * blockDim.x + threadIdx.x;
  if (i < n) P[i] = silu_f(P[i]);
}
__global__ void gelu_inplace_kernel(float* __restrict__ P, size_t n) {
  size_t i = (size_t)blockIdx.x * blockDim.x + threadIdx.x;
  if (i < n) P[i] = gelu_f(P[i]);
}
__global__ void slice_kernel(const float* __restrict__ S, int srcC, int srcOff,
                             float* __restrict__ D, int dstC, size_t rows) {
  size_t i = (size_t)blockIdx.x * blockDim.x + threadIdx.x;
  size_t total = rows * dstC;
  if (i >= total) return;
  size_t r = i / dstC; int c = (int)(i % dstC);
  D[i] = S[r * srcC + srcOff + c];
}
// out[r,0:hid] = gelu(src[r,0:hid]) * src[r,hid:2hid]
__global__ void gelugate_kernel(const float* __restrict__ S, float* __restrict__ D,
                                size_t rows, int hid) {
  size_t i = (size_t)blockIdx.x * blockDim.x + threadIdx.x;
  size_t total = rows * hid;
  if (i >= total) return;
  size_t r = i / hid; int c = (int)(i % hid);
  const float a = S[r * 2 * hid + c];
  D[i] = gelu_f(a) * S[r * 2 * hid + hid + c];
}
// ym[r,d] *= silu(z[r*256 + 128 + d])
__global__ void zgate_kernel(float* __restrict__ YM, const float* __restrict__ Z, size_t rows) {
  size_t i = (size_t)blockIdx.x * blockDim.x + threadIdx.x;
  size_t total = rows * 128;
  if (i >= total) return;
  size_t r = i / 128; int d = (int)(i % 128);
  YM[i] *= silu_f(Z[r * 256 + 128 + d]);
}

// ---------------------------------------------------------------------------
// q/k L2 normalize over L per (b, channel), in place
// ---------------------------------------------------------------------------
__global__ void l2norm_kernel(float* __restrict__ Q, int L, int C) {
  const int c = blockIdx.x, b = blockIdx.y;
  __shared__ float red[256];
  float s = 0.f;
  for (int l = threadIdx.x; l < L; l += blockDim.x) {
    float v = Q[((size_t)b * L + l) * C + c]; s += v * v;
  }
  red[threadIdx.x] = s; __syncthreads();
  for (int st = blockDim.x >> 1; st; st >>= 1) {
    if ((int)threadIdx.x < st) red[threadIdx.x] += red[threadIdx.x + st];
    __syncthreads();
  }
  const float inv = 1.f / (sqrtf(red[0]) + 1e-12f);
  for (int l = threadIdx.x; l < L; l += blockDim.x)
    Q[((size_t)b * L + l) * C + c] *= inv;
}

// attn[b,h,c,d] = temp[h] * sum_l q[b,l,h*8+c] * k[b,l,h*8+d]
__global__ void attn_logits_kernel(const float* __restrict__ Q, const float* __restrict__ Kb,
                                   const float* __restrict__ temp, float* __restrict__ A,
                                   int L, int C) {
  const int cd = blockIdx.x, hh = blockIdx.y, b = blockIdx.z;
  const int c = cd >> 3, d = cd & 7;
  const int cq = hh * 8 + c, ck = hh * 8 + d;
  __shared__ float red[128];
  float s = 0.f;
  for (int l = threadIdx.x; l < L; l += blockDim.x)
    s += Q[((size_t)b * L + l) * C + cq] * Kb[((size_t)b * L + l) * C + ck];
  red[threadIdx.x] = s; __syncthreads();
  for (int st = blockDim.x >> 1; st; st >>= 1) {
    if ((int)threadIdx.x < st) red[threadIdx.x] += red[threadIdx.x + st];
    __syncthreads();
  }
  if (threadIdx.x == 0) A[(((size_t)b * 8 + hh) * 8 + c) * 8 + d] = red[0] * temp[hh];
}

__global__ void softmax8_kernel(float* __restrict__ A, int rows) {
  int i = blockIdx.x * blockDim.x + threadIdx.x;
  if (i >= rows) return;
  float* r = A + (size_t)i * 8;
  float mx = r[0];
#pragma unroll
  for (int d = 1; d < 8; ++d) mx = fmaxf(mx, r[d]);
  float s = 0.f;
#pragma unroll
  for (int d = 0; d < 8; ++d) { r[d] = __expf(r[d] - mx); s += r[d]; }
  const float inv = 1.f / s;
#pragma unroll
  for (int d = 0; d < 8; ++d) r[d] *= inv;
}

// out[b,l,h*8+c] = sum_d attn[b,h,c,d] * v[b,l,h*8+d]
__global__ void attn_apply_kernel(const float* __restrict__ A, const float* __restrict__ V,
                                  float* __restrict__ O, size_t total, int L, int C) {
  size_t i = (size_t)blockIdx.x * blockDim.x + threadIdx.x;
  if (i >= total) return;
  size_t r = i / C; int cf = (int)(i % C);
  const int b = (int)(r / L);
  const int hh = cf >> 3, c = cf & 7;
  const float* ar = A + (((size_t)b * 8 + hh) * 8 + c) * 8;
  const float* vr = V + r * C + hh * 8;
  float s = 0.f;
#pragma unroll
  for (int d = 0; d < 8; ++d) s += ar[d] * vr[d];
  O[i] = s;
}

// ---------------------------------------------------------------------------
// SS2D: delta = softplus(xdbl[:, :4] . dtw + dtb)
// ---------------------------------------------------------------------------
__global__ void dtdelta_kernel(const float* __restrict__ XD, const float* __restrict__ DTW,
                               const float* __restrict__ DTB, float* __restrict__ DEL,
                               size_t BL) {
  size_t i = (size_t)blockIdx.x * blockDim.x + threadIdx.x;
  size_t total = 4 * BL * 128;
  if (i >= total) return;
  const size_t row = i / 128;
  const int d = (int)(i % 128);
  const int k = (int)(row / BL);
  const float* xr = XD + row * 36;
  const float* wr = DTW + ((size_t)(k * 128 + d)) * 4;
  float acc = DTB[k * 128 + d];
#pragma unroll
  for (int rr = 0; rr < 4; ++rr) acc += xr[rr] * wr[rr];
  DEL[i] = softplus_f(acc);
}

DEVFN int scan_pos(int k, int s, int H, int W, int L) {
  int t = (k >= 2) ? (L - 1 - s) : s;
  if ((k & 1) == 0) return t;          // row-major scan
  const int jw = t / H, ih = t % H;    // (w,h)-order scan
  return ih * W + jw;
}

// One block of 128 threads per (k,b): thread = channel d, 16-state in regs.
// B/C slices of each xdbl row are 16B-aligned -> float4 broadcast loads.
__global__ void scan_kernel(const float* __restrict__ XC,   // [B,L,128] spatial
                            const float* __restrict__ XD,   // [K,B,L,36] spatial
                            const float* __restrict__ DEL,  // [K,B,L,128] spatial
                            const float* __restrict__ ALOG, // [512,16]
                            float* __restrict__ OY,         // [K,B,L,128] spatial
                            int Bn, int L, int H, int W) {
  const int d = threadIdx.x;      // 0..127
  const int k = blockIdx.x;       // 0..3
  const int b = blockIdx.y;
  float hs[16], Av[16];
#pragma unroll
  for (int n = 0; n < 16; ++n) {
    hs[n] = 0.f;
    Av[n] = -__expf(ALOG[((size_t)(k * 128 + d)) * 16 + n]);
  }
  const size_t base = ((size_t)k * Bn + b) * (size_t)L;
  const float* dl_base = DEL + base * 128;
  const float* xd_base = XD + base * 36;
  float* oy_base = OY + base * 128;
  const float* xc_base = XC + (size_t)b * L * 128;
  for (int s = 0; s < L; ++s) {
    const int pos = scan_pos(k, s, H, W, L);
    const float dl = dl_base[(size_t)pos * 128 + d];
    const float u  = xc_base[(size_t)pos * 128 + d];
    const float du = dl * u;
    const float4* bv = (const float4*)(xd_base + (size_t)pos * 36 + 4);
    const float4* cv = (const float4*)(xd_base + (size_t)pos * 36 + 20);
    float Bv[16], Cv[16];
#pragma unroll
    for (int q = 0; q < 4; ++q) {
      float4 tb = bv[q]; Bv[4*q] = tb.x; Bv[4*q+1] = tb.y; Bv[4*q+2] = tb.z; Bv[4*q+3] = tb.w;
      float4 tc = cv[q]; Cv[4*q] = tc.x; Cv[4*q+1] = tc.y; Cv[4*q+2] = tc.z; Cv[4*q+3] = tc.w;
    }
    float y = 0.f;
#pragma unroll
    for (int n = 0; n < 16; ++n) {
      hs[n] = __expf(dl * Av[n]) * hs[n] + du * Bv[n];
      y += hs[n] * Cv[n];
    }
    oy_base[(size_t)pos * 128 + d] = y;
  }
}

// ym[r,d] = sum_k oy[k*BL + r, d] + (sum_k D[k,d]) * xc[r,d]
__global__ void merge_kernel(const float* __restrict__ OY, const float* __restrict__ Dp,
                             const float* __restrict__ XC, float* __restrict__ YM, size_t BL) {
  size_t i = (size_t)blockIdx.x * blockDim.x + threadIdx.x;
  size_t total = BL * 128;
  if (i >= total) return;
  size_t r = i / 128; int d = (int)(i % 128);
  float s = 0.f;
#pragma unroll
  for (int k = 0; k < 4; ++k) s += OY[((size_t)k * BL + r) * 128 + d];
  const float Ds = Dp[d] + Dp[128 + d] + Dp[256 + d] + Dp[384 + d];
  YM[i] = s + Ds * XC[i];
}

// ---------------------------------------------------------------------------
// Host driver
// ---------------------------------------------------------------------------
static inline unsigned ew_blocks(size_t n) { return (unsigned)((n + 255) / 256); }

extern "C" void kernel_launch(void* const* d_in, const int* in_sizes, int n_in,
                              void* d_out, int out_size, void* d_ws, size_t ws_size,
                              hipStream_t stream) {
  (void)n_in; (void)out_size; (void)ws_size;
  const int C = 64, Hh = 64, Ww = 64, HID = 170;
  const int L = Hh * Ww;
  const int Bn = in_sizes[0] / (L * C);       // 4
  const size_t BL = (size_t)Bn * L;

  float* ws = (float*)d_ws;
  size_t off = 0;
  auto alloc = [&](size_t n) { float* p = ws + off; off += n; return p; };
  float* bufA    = alloc(BL * 64);
  float* bufB    = alloc(BL * 64);
  float* bufQ    = alloc(BL * 64);
  float* bufV    = alloc(BL * 64);
  float* bufY    = alloc(BL * 64);
  float* buf128a = alloc(BL * 128);
  float* bufXC   = alloc(BL * 128);
  float* buf256  = alloc(BL * 256);
  float* buf340a = alloc(BL * 340);   // also reused as FF 256-wide temp
  float* buf340b = alloc(BL * 340);
  float* buf170  = alloc(BL * 170);
  float* attnb   = alloc((size_t)Bn * 8 * 8 * 8);
  float* xdbl    = alloc(4 * BL * 36);
  float* delta   = alloc(4 * BL * 128);
  float* outy    = alloc(4 * BL * 128);
  float* ym      = alloc(BL * 128);

  float* xcur = (float*)d_out;   // running tensor, [B, L, C] channel-last

  auto gemm = [&](const float* X, const float* W, const float* bias, float* Y,
                  int M, int N, int K, int acc) {
    dim3 g((N + 31) / 32, M / 64);
    gemm16_wmma<<<g, dim3(128), 0, stream>>>(X, W, bias, Y, M, N, K, acc);
  };
  auto ln = [&](const float* X, const float* g, const float* b, float* Y, int Cc) {
    ln_kernel<<<dim3((unsigned)BL), dim3(32), 0, stream>>>(X, g, b, Y, Cc);
  };
  auto dwconv = [&](const float* X, const float* W, const float* bias, float* Y, int Cc) {
    size_t n = BL * Cc;
    dwconv3_kernel<<<ew_blocks(n), 256, 0, stream>>>(X, W, bias, Y, Bn, Hh, Ww, Cc);
  };

  copy_kernel<<<ew_blocks(BL * 64), 256, 0, stream>>>((const float*)d_in[0], xcur, BL * 64);

  for (int blk = 0; blk < 2; ++blk) {
    auto p = [&](int i) { return (const float*)d_in[1 + blk * 39 + i]; };
    // ifa params
    const float *n1g = p(0), *n1b = p(1), *n2g = p(2), *n2b = p(3);
    const float *qw = p(4), *qb = p(5), *qdw = p(6), *qdwb = p(7);
    const float *kvw = p(8), *kvb = p(9), *kvdw = p(10), *kvdwb = p(11);
    const float *temp = p(12), *pw = p(13), *pb = p(14), *n3g = p(15), *n3b = p(16);
    const float *fiw = p(17), *fib = p(18), *fdw = p(19), *fdb = p(20);
    const float *fow = p(21), *fob = p(22);
    // ss2d params
    const float *inw = p(23), *cw = p(24), *cb = p(25), *xw = p(26);
    const float *dtw = p(27), *dtb = p(28), *Alog = p(29), *Dp = p(30);
    const float *ong = p(31), *onb = p(32), *oww = p(33);
    // ff params
    const float *ng = p(34), *nb = p(35), *w1 = p(36), *wd = p(37), *w2 = p(38);

    // ---------------- IFA (y = ifa(x,x)) ----------------
    copy_kernel<<<ew_blocks(BL * 64), 256, 0, stream>>>(xcur, bufY, BL * 64);
    ln(xcur, n1g, n1b, bufA, 64);
    gemm(bufA, qw, qb, bufB, (int)BL, 64, 64, 0);
    dwconv(bufB, qdw, qdwb, bufQ, 64);
    ln(xcur, n2g, n2b, bufA, 64);
    gemm(bufA, kvw, kvb, buf128a, (int)BL, 128, 64, 0);
    dwconv(buf128a, kvdw, kvdwb, buf256, 128);   // buf256 front half as kv temp
    slice_kernel<<<ew_blocks(BL * 64), 256, 0, stream>>>(buf256, 128, 0,  bufB, 64, BL);
    slice_kernel<<<ew_blocks(BL * 64), 256, 0, stream>>>(buf256, 128, 64, bufV, 64, BL);
    l2norm_kernel<<<dim3(64, Bn), 256, 0, stream>>>(bufQ, L, 64);
    l2norm_kernel<<<dim3(64, Bn), 256, 0, stream>>>(bufB, L, 64);
    attn_logits_kernel<<<dim3(64, 8, Bn), 128, 0, stream>>>(bufQ, bufB, temp, attnb, L, 64);
    softmax8_kernel<<<ew_blocks((size_t)Bn * 64), 256, 0, stream>>>(attnb, Bn * 64);
    attn_apply_kernel<<<ew_blocks(BL * 64), 256, 0, stream>>>(attnb, bufV, bufA, BL * 64, L, 64);
    gemm(bufA, pw, pb, bufY, (int)BL, 64, 64, 1);          // y = x + proj(attn)
    ln(bufY, n3g, n3b, bufA, 64);
    gemm(bufA, fiw, fib, buf340a, (int)BL, 340, 64, 0);
    dwconv(buf340a, fdw, fdb, buf340b, 340);
    gelugate_kernel<<<ew_blocks(BL * HID), 256, 0, stream>>>(buf340b, buf170, BL, HID);
    gemm(buf170, fow, fob, bufY, (int)BL, 64, HID, 1);     // y += ffn-gate

    // ---------------- SS2D(y) + x ----------------
    gemm(bufY, inw, nullptr, buf256, (int)BL, 256, 64, 0);
    slice_kernel<<<ew_blocks(BL * 128), 256, 0, stream>>>(buf256, 256, 0, buf128a, 128, BL);
    dwconv(buf128a, cw, cb, bufXC, 128);
    silu_inplace_kernel<<<ew_blocks(BL * 128), 256, 0, stream>>>(bufXC, BL * 128);
    for (int k = 0; k < 4; ++k)
      gemm(bufXC, xw + (size_t)k * 36 * 128, nullptr, xdbl + (size_t)k * BL * 36,
           (int)BL, 36, 128, 0);
    dtdelta_kernel<<<ew_blocks(4 * BL * 128), 256, 0, stream>>>(xdbl, dtw, dtb, delta, BL);
    scan_kernel<<<dim3(4, Bn), 128, 0, stream>>>(bufXC, xdbl, delta, Alog, outy, Bn, L, Hh, Ww);
    merge_kernel<<<ew_blocks(BL * 128), 256, 0, stream>>>(outy, Dp, bufXC, ym, BL);
    ln(ym, ong, onb, ym, 128);
    zgate_kernel<<<ew_blocks(BL * 128), 256, 0, stream>>>(ym, buf256, BL);
    gemm(ym, oww, nullptr, xcur, (int)BL, 64, 128, 1);     // x = ss2d(y) + x

    // ---------------- FF ----------------
    ln(xcur, ng, nb, bufA, 64);
    gemm(bufA, w1, nullptr, buf256, (int)BL, 256, 64, 0);
    gelu_inplace_kernel<<<ew_blocks(BL * 256), 256, 0, stream>>>(buf256, BL * 256);
    dwconv(buf256, wd, nullptr, buf340a, 256);             // reuse 340 buffer as 256 temp
    gelu_inplace_kernel<<<ew_blocks(BL * 256), 256, 0, stream>>>(buf340a, BL * 256);
    gemm(buf340a, w2, nullptr, xcur, (int)BL, 64, 256, 1); // x = ff(x) + x
  }
}
